// LagunaMoE_36369783062549
// MI455X (gfx1250) — matrix operand order repaired and checked by reading, not compiled, not run on hardware
//
#include <hip/hip_runtime.h>
#include <hip/hip_bf16.h>
#include <math.h>

// ---------------- problem constants ----------------
constexpr int T  = 2048;
constexpr int H  = 2048;
constexpr int E  = 16;
constexpr int KT = 4;          // top_k
constexpr int I  = 1408;       // routed intermediate
constexpr int IS = 5632;       // shared intermediate
constexpr float SCALE = 2.5f;
constexpr float CAP   = 30.0f;
constexpr int NSLOTS = T * KT; // 8192

// ---------------- WMMA types ----------------
typedef __attribute__((ext_vector_type(16))) unsigned short v16u;
typedef __attribute__((ext_vector_type(16))) __bf16        v16bf;
typedef __attribute__((ext_vector_type(8)))  float          v8f;
union FragU { v16u u; v16bf b; };

#define ASTR 40   // LDS row stride (ushorts): 80B, multiple of 16B
#define BSTR 40

// ---------------- gfx1250 async global->LDS copy ----------------
typedef int v4i __attribute__((__vector_size__(16)));

#if defined(__has_builtin)
#if __has_builtin(__builtin_amdgcn_global_load_async_to_lds_b128)
#define HAVE_ASYNC_B128 1
#endif
#if __has_builtin(__builtin_amdgcn_s_wait_asynccnt)
#define HAVE_WAIT_ASYNC 1
#endif
#endif

#ifdef HAVE_ASYNC_B128
#define ASYNC_COPY_B128(g, l)                                                  \
  __builtin_amdgcn_global_load_async_to_lds_b128(                              \
      (__attribute__((address_space(1))) v4i*)(g),                             \
      (__attribute__((address_space(3))) v4i*)(l), 0, 0)
#else
#define ASYNC_COPY_B128(g, l)                                                  \
  asm volatile("global_load_async_to_lds_b128 %0, %1, off"                     \
               :: "v"((unsigned)(unsigned long long)(l)),                      \
                  "v"((unsigned long long)(g)) : "memory")
#endif

#ifdef HAVE_WAIT_ASYNC
#define WAIT_ASYNC() __builtin_amdgcn_s_wait_asynccnt(0)
#else
#define WAIT_ASYNC() asm volatile("s_wait_asynccnt 0x0" ::: "memory")
#endif

__device__ __forceinline__ unsigned short f2bf(float f) {
  unsigned int u = __builtin_bit_cast(unsigned int, f);
  u += 0x7FFFu + ((u >> 16) & 1u);            // round-to-nearest-even
  return (unsigned short)(u >> 16);
}
__device__ __forceinline__ unsigned pack2bf(float f0, float f1) {
  return (unsigned)f2bf(f0) | ((unsigned)f2bf(f1) << 16);
}
__device__ __forceinline__ float fast_rcp(float x) { return __builtin_amdgcn_rcpf(x); }
__device__ __forceinline__ float silu(float g) {
  return g * fast_rcp(1.0f + __expf(-g));     // v_rcp_f32, not IEEE div chain
}

// A fragment (16x32 bf16): lane<16 -> row=lane, K {0..7}U{16..23};
//                          lane>=16 -> row=lane-16, K {8..15}U{24..31}
__device__ __forceinline__ v16u ld_a_frag(const unsigned short* base, int lane) {
  int row = lane & 15;
  int k0  = (lane & 16) ? 8 : 0;
  const unsigned short* p = base + row * ASTR + k0;
  v16u r;
  ((uint4*)&r)[0] = *(const uint4*)(p);
  ((uint4*)&r)[1] = *(const uint4*)(p + 16);
  return r;
}

// B fragment (32x16 bf16, staged K-contiguous per column in LDS):
// lane<16 -> col=lane, K {0..7}U{16..23} ; lane>=16 -> col=lane-16, K {8..15}U{24..31}
__device__ __forceinline__ v16u ld_b_frag(const unsigned short* base, int lane) {
  int col = lane & 15;
  int k0  = (lane & 16) ? 8 : 0;
  const unsigned short* p = base + col * BSTR + k0;
  v16u r;
  ((uint4*)&r)[0] = *(const uint4*)(p);
  ((uint4*)&r)[1] = *(const uint4*)(p + 16);
  return r;
}

// stage 128x32 bf16 A tile from gathered rows (tokLds): async global->LDS DMA
__device__ __forceinline__ void stage_A(unsigned short* As, const unsigned short* src,
                                        int ld, const int* tokLds, int kbase, int tid) {
#pragma unroll
  for (int i = 0; i < 2; i++) {
    int seg = tid + i * 256;
    int row = seg >> 2;
    int kc  = (seg & 3) * 8;
    int tok = tokLds[row];
    const unsigned short* g = src + (size_t)tok * ld + kbase + kc;
    unsigned short* l = As + row * ASTR + kc;
    ASYNC_COPY_B128(g, l);
  }
}

// stage 32x64 fp32 weight tile -> bf16 transposed in LDS (Bst[n][k]).
// Each thread converts two adjacent-K values for one column -> packed b32 store.
__device__ __forceinline__ void stage_B(unsigned short* Bst, const float* W, int ldw,
                                        int kbase, int n0, int tid) {
  int n  = tid & 63;
  int kp = tid >> 6;                          // 0..3
  const float* base = W + (size_t)kbase * ldw + n0 + n;
  __builtin_prefetch(base + (size_t)32 * ldw, 0, 3);   // next K-slab -> global_prefetch
#pragma unroll
  for (int kk = 0; kk < 4; kk++) {
    int k = (kp + kk * 4) * 2;
    float f0 = base[(size_t)k * ldw];
    float f1 = base[(size_t)(k + 1) * ldw];
    *(unsigned*)(Bst + n * BSTR + k) = pack2bf(f0, f1);
  }
}

__device__ __forceinline__ void mma_step2(const unsigned short* As,
                                          const unsigned short* Bg,
                                          const unsigned short* Bu,
                                          int wave, int lane, v8f* accg, v8f* accu) {
  FragU a; a.u = ld_a_frag(As + wave * 16 * ASTR, lane);
#pragma unroll
  for (int j = 0; j < 4; j++) {
    FragU bg; bg.u = ld_b_frag(Bg + j * 16 * BSTR, lane);
    accg[j] = __builtin_amdgcn_wmma_f32_16x16x32_bf16(false, a.b, false, bg.b,
                                                      (short)0, accg[j], false, false);
    FragU bu; bu.u = ld_b_frag(Bu + j * 16 * BSTR, lane);
    accu[j] = __builtin_amdgcn_wmma_f32_16x16x32_bf16(false, a.b, false, bu.b,
                                                      (short)0, accu[j], false, false);
  }
}

__device__ __forceinline__ void mma_step1(const unsigned short* As,
                                          const unsigned short* Bs,
                                          int wave, int lane, v8f* acc) {
  FragU a; a.u = ld_a_frag(As + wave * 16 * ASTR, lane);
#pragma unroll
  for (int j = 0; j < 4; j++) {
    FragU b; b.u = ld_b_frag(Bs + j * 16 * BSTR, lane);
    acc[j] = __builtin_amdgcn_wmma_f32_16x16x32_bf16(false, a.b, false, b.b,
                                                     (short)0, acc[j], false, false);
  }
}

// ---------------- small kernels ----------------
__global__ void zero_small(int* counts, int* fill) {
  int t = threadIdx.x;
  if (t < E) { counts[t] = 0; fill[t] = 0; }
}

__global__ void cast_x(const float* __restrict__ x, unsigned short* __restrict__ xb) {
  size_t i = ((size_t)blockIdx.x * 256 + threadIdx.x) * 8;
  float4 a = *(const float4*)(x + i);
  float4 b = *(const float4*)(x + i + 4);
  uint4 o;
  o.x = pack2bf(a.x, a.y);
  o.y = pack2bf(a.z, a.w);
  o.z = pack2bf(b.x, b.y);
  o.w = pack2bf(b.z, b.w);
  *(uint4*)(xb + i) = o;
}

// one wave32 per token: logits -> tanh softcap -> sigmoid -> top4 -> renorm
__global__ __launch_bounds__(256)
void router_kernel(const float* __restrict__ x, const float* __restrict__ gw,
                   const float* __restrict__ bias, int* __restrict__ topk_ids,
                   float* __restrict__ topk_w, int* __restrict__ counts) {
  int wave = threadIdx.x >> 5, lane = threadIdx.x & 31;
  int t = blockIdx.x * 8 + wave;
  const float* xr = x + (size_t)t * H;
  float acc[E];
#pragma unroll
  for (int e = 0; e < E; e++) acc[e] = 0.0f;
  for (int h = lane; h < H; h += 32) {
    float xv = xr[h];
#pragma unroll
    for (int e = 0; e < E; e++) acc[e] += xv * gw[e * H + h];
  }
#pragma unroll
  for (int e = 0; e < E; e++)
    for (int off = 16; off > 0; off >>= 1) acc[e] += __shfl_xor(acc[e], off);

  if (lane == 0) {
    float sc[E], sel[E];
    bool taken[E];
#pragma unroll
    for (int e = 0; e < E; e++) {
      float l = tanhf(acc[e] / CAP) * CAP;
      float s = fast_rcp(1.0f + __expf(-l));
      sc[e] = s;
      sel[e] = s + bias[e];
      taken[e] = false;
    }
    int ids[KT]; float ws[KT]; float sum = 0.0f;
#pragma unroll
    for (int k = 0; k < KT; k++) {
      int best = 0; float bv = -1e30f;
      for (int e = 0; e < E; e++)
        if (!taken[e] && sel[e] > bv) { bv = sel[e]; best = e; }
      taken[best] = true;
      ids[k] = best;
      ws[k] = sc[best];
      sum += sc[best];
    }
    float inv = fast_rcp(sum);
#pragma unroll
    for (int k = 0; k < KT; k++) {
      topk_ids[t * KT + k] = ids[k];
      topk_w[t * KT + k]   = ws[k] * inv;
      atomicAdd(&counts[ids[k]], 1);
    }
  }
}

__global__ void prefix_kernel(const int* counts, int* offs) {
  if (threadIdx.x == 0) {
    int s = 0;
    for (int e = 0; e < E; e++) { offs[e] = s; s += counts[e]; }
    offs[E] = s;
  }
}

__global__ void scatter_kernel(const int* __restrict__ topk_ids,
                               const float* __restrict__ topk_w,
                               const int* __restrict__ offs, int* __restrict__ fill,
                               int* __restrict__ bucket_tok,
                               float* __restrict__ bucket_w) {
  int t = blockIdx.x * 256 + threadIdx.x;
#pragma unroll
  for (int k = 0; k < KT; k++) {
    int e = topk_ids[t * KT + k];
    int pos = atomicAdd(&fill[e], 1);
    int slot = offs[e] + pos;
    bucket_tok[slot] = t;
    bucket_w[slot]   = topk_w[t * KT + k] * SCALE;  // fold routed_scaling_factor
  }
}

// ---------------- shared expert ----------------
// act = silu(x @ Wg) * (x @ Wu), W = [H, 2*IS] row-major
__global__ __launch_bounds__(256)
void shared_gemm1(const unsigned short* __restrict__ xb, const float* __restrict__ W,
                  unsigned short* __restrict__ act) {
  __shared__ __align__(16) unsigned short As[128 * ASTR];
  __shared__ __align__(16) unsigned short Bg[64 * BSTR];
  __shared__ __align__(16) unsigned short Bu[64 * BSTR];
  __shared__ int tokLds[128];
  int tid = threadIdx.x, wave = tid >> 5, lane = tid & 31;
  int n0 = blockIdx.x * 64;
  int m0 = blockIdx.y * 128;
  if (tid < 128) tokLds[tid] = m0 + tid;
  v8f accg[4], accu[4];
#pragma unroll
  for (int j = 0; j < 4; j++)
    for (int r = 0; r < 8; r++) { accg[j][r] = 0.0f; accu[j][r] = 0.0f; }
  __syncthreads();
  for (int kb = 0; kb < H; kb += 32) {
    stage_A(As, xb, H, tokLds, kb, tid);
    stage_B(Bg, W, 2 * IS, kb, n0, tid);
    stage_B(Bu, W, 2 * IS, kb, IS + n0, tid);
    WAIT_ASYNC();
    __syncthreads();
    mma_step2(As, Bg, Bu, wave, lane, accg, accu);
    __syncthreads();
  }
  int mb  = (lane & 16) ? 8 : 0;
  int nl  = lane & 15;
#pragma unroll
  for (int j = 0; j < 4; j++) {
    int c = n0 + j * 16 + nl;
#pragma unroll
    for (int r = 0; r < 8; r++) {
      int m = m0 + wave * 16 + mb + r;
      act[(size_t)m * IS + c] = f2bf(silu(accg[j][r]) * accu[j][r]);
    }
  }
}

// out = act @ Wd, Wd = [IS, H] row-major, plain fp32 stores (initializes d_out)
__global__ __launch_bounds__(256)
void shared_gemm2(const unsigned short* __restrict__ act, const float* __restrict__ Wd,
                  float* __restrict__ out) {
  __shared__ __align__(16) unsigned short As[128 * ASTR];
  __shared__ __align__(16) unsigned short Bs[64 * BSTR];
  __shared__ int tokLds[128];
  int tid = threadIdx.x, wave = tid >> 5, lane = tid & 31;
  int n0 = blockIdx.x * 64;
  int m0 = blockIdx.y * 128;
  if (tid < 128) tokLds[tid] = m0 + tid;
  v8f acc[4];
#pragma unroll
  for (int j = 0; j < 4; j++)
    for (int r = 0; r < 8; r++) acc[j][r] = 0.0f;
  __syncthreads();
  for (int kb = 0; kb < IS; kb += 32) {
    stage_A(As, act, IS, tokLds, kb, tid);
    stage_B(Bs, Wd, H, kb, n0, tid);
    WAIT_ASYNC();
    __syncthreads();
    mma_step1(As, Bs, wave, lane, acc);
    __syncthreads();
  }
  int mb = (lane & 16) ? 8 : 0;
  int nl = lane & 15;
#pragma unroll
  for (int j = 0; j < 4; j++) {
    int c = n0 + j * 16 + nl;
#pragma unroll
    for (int r = 0; r < 8; r++) {
      int m = m0 + wave * 16 + mb + r;
      out[(size_t)m * H + c] = acc[j][r];
    }
  }
}

// ---------------- routed experts (grouped GEMM over buckets) ----------------
__global__ __launch_bounds__(256)
void routed_gemm1(const unsigned short* __restrict__ xb, const float* __restrict__ Wge,
                  const int* __restrict__ counts, const int* __restrict__ offs,
                  const int* __restrict__ bucket_tok, unsigned short* __restrict__ act) {
  int e = blockIdx.z;
  int count = counts[e];
  int m0 = blockIdx.y * 128;
  if (m0 >= count) return;
  const float* W = Wge + (size_t)e * H * (2 * I);
  int slot0 = offs[e] + m0;

  __shared__ __align__(16) unsigned short As[128 * ASTR];
  __shared__ __align__(16) unsigned short Bg[64 * BSTR];
  __shared__ __align__(16) unsigned short Bu[64 * BSTR];
  __shared__ int tokLds[128];
  int tid = threadIdx.x, wave = tid >> 5, lane = tid & 31;
  int n0 = blockIdx.x * 64;
  if (tid < 128)
    tokLds[tid] = (m0 + tid < count) ? bucket_tok[slot0 + tid] : 0;
  v8f accg[4], accu[4];
#pragma unroll
  for (int j = 0; j < 4; j++)
    for (int r = 0; r < 8; r++) { accg[j][r] = 0.0f; accu[j][r] = 0.0f; }
  __syncthreads();
  for (int kb = 0; kb < H; kb += 32) {
    stage_A(As, xb, H, tokLds, kb, tid);
    stage_B(Bg, W, 2 * I, kb, n0, tid);
    stage_B(Bu, W, 2 * I, kb, I + n0, tid);
    WAIT_ASYNC();
    __syncthreads();
    mma_step2(As, Bg, Bu, wave, lane, accg, accu);
    __syncthreads();
  }
  int mb = (lane & 16) ? 8 : 0;
  int nl = lane & 15;
#pragma unroll
  for (int j = 0; j < 4; j++) {
    int c = n0 + j * 16 + nl;
#pragma unroll
    for (int r = 0; r < 8; r++) {
      int ml = wave * 16 + mb + r;
      if (m0 + ml < count)
        act[(size_t)(slot0 + ml) * I + c] = f2bf(silu(accg[j][r]) * accu[j][r]);
    }
  }
}

__global__ __launch_bounds__(256)
void routed_gemm2(const unsigned short* __restrict__ act, const float* __restrict__ Wde,
                  const int* __restrict__ counts, const int* __restrict__ offs,
                  const int* __restrict__ bucket_tok, const float* __restrict__ bucket_w,
                  float* __restrict__ out) {
  int e = blockIdx.z;
  int count = counts[e];
  int m0 = blockIdx.y * 128;
  if (m0 >= count) return;
  const float* W = Wde + (size_t)e * I * H;
  int slot0 = offs[e] + m0;

  __shared__ __align__(16) unsigned short As[128 * ASTR];
  __shared__ __align__(16) unsigned short Bs[64 * BSTR];
  __shared__ int rowLds[128];
  __shared__ int rowTok[128];
  __shared__ float rowW[128];
  int tid = threadIdx.x, wave = tid >> 5, lane = tid & 31;
  int n0 = blockIdx.x * 64;
  if (tid < 128) {
    int s = slot0 + tid;
    if (s > NSLOTS - 1) s = NSLOTS - 1;
    rowLds[tid] = s;                       // contiguous slot rows (A has no gather)
    if (m0 + tid < count) { rowTok[tid] = bucket_tok[slot0 + tid]; rowW[tid] = bucket_w[slot0 + tid]; }
    else                  { rowTok[tid] = 0; rowW[tid] = 0.0f; }
  }
  v8f acc[4];
#pragma unroll
  for (int j = 0; j < 4; j++)
    for (int r = 0; r < 8; r++) acc[j][r] = 0.0f;
  __syncthreads();
  for (int kb = 0; kb < I; kb += 32) {
    stage_A(As, act, I, rowLds, kb, tid);
    stage_B(Bs, W, H, kb, n0, tid);
    WAIT_ASYNC();
    __syncthreads();
    mma_step1(As, Bs, wave, lane, acc);
    __syncthreads();
  }
  int mb = (lane & 16) ? 8 : 0;
  int nl = lane & 15;
#pragma unroll
  for (int j = 0; j < 4; j++) {
    int c = n0 + j * 16 + nl;
#pragma unroll
    for (int r = 0; r < 8; r++) {
      int ml = wave * 16 + mb + r;
      if (m0 + ml < count) {
        float v = acc[j][r] * rowW[ml];
        atomicAdd(&out[(size_t)rowTok[ml] * H + c], v);
      }
    }
  }
}

// ---------------- workspace layout ----------------
constexpr size_t ALN = 256;
constexpr size_t al(size_t x) { return (x + ALN - 1) & ~(ALN - 1); }
constexpr size_t OFF_XB   = 0;
constexpr size_t OFF_ACTS = OFF_XB   + al((size_t)T * H * 2);
constexpr size_t OFF_ACTR = OFF_ACTS + al((size_t)T * IS * 2);
constexpr size_t OFF_TID  = OFF_ACTR + al((size_t)NSLOTS * I * 2);
constexpr size_t OFF_TW   = OFF_TID  + al((size_t)T * KT * 4);
constexpr size_t OFF_BTOK = OFF_TW   + al((size_t)T * KT * 4);
constexpr size_t OFF_BW   = OFF_BTOK + al((size_t)NSLOTS * 4);
constexpr size_t OFF_CNT  = OFF_BW   + al((size_t)NSLOTS * 4);
constexpr size_t OFF_FILL = OFF_CNT  + al((size_t)E * 4);
constexpr size_t OFF_OFFS = OFF_FILL + al((size_t)E * 4);

extern "C" void kernel_launch(void* const* d_in, const int* in_sizes, int n_in,
                              void* d_out, int out_size, void* d_ws, size_t ws_size,
                              hipStream_t stream) {
  const float* x      = (const float*)d_in[0];
  const float* gate_w = (const float*)d_in[1];
  const float* bias   = (const float*)d_in[2];
  const float* w_gu   = (const float*)d_in[3];
  const float* w_dn   = (const float*)d_in[4];
  const float* sh_gu  = (const float*)d_in[5];
  const float* sh_dn  = (const float*)d_in[6];
  float* out = (float*)d_out;

  char* ws = (char*)d_ws;
  unsigned short* xb   = (unsigned short*)(ws + OFF_XB);
  unsigned short* actS = (unsigned short*)(ws + OFF_ACTS);
  unsigned short* actR = (unsigned short*)(ws + OFF_ACTR);
  int*   topk_ids   = (int*)  (ws + OFF_TID);
  float* topk_w     = (float*)(ws + OFF_TW);
  int*   bucket_tok = (int*)  (ws + OFF_BTOK);
  float* bucket_w   = (float*)(ws + OFF_BW);
  int*   counts     = (int*)  (ws + OFF_CNT);
  int*   fill       = (int*)  (ws + OFF_FILL);
  int*   offs       = (int*)  (ws + OFF_OFFS);

  zero_small<<<1, 64, 0, stream>>>(counts, fill);
  cast_x<<<(T * H) / (256 * 8), 256, 0, stream>>>(x, xb);
  router_kernel<<<T / 8, 256, 0, stream>>>(x, gate_w, bias, topk_ids, topk_w, counts);
  prefix_kernel<<<1, 32, 0, stream>>>(counts, offs);
  scatter_kernel<<<T / 256, 256, 0, stream>>>(topk_ids, topk_w, offs, fill,
                                              bucket_tok, bucket_w);
  // shared expert
  shared_gemm1<<<dim3(IS / 64, T / 128), 256, 0, stream>>>(xb, sh_gu, actS);
  shared_gemm2<<<dim3(H / 64, T / 128), 256, 0, stream>>>(actS, sh_dn, out);
  // routed experts (grouped over buckets); atomics accumulate after shared write
  routed_gemm1<<<dim3(I / 64, T / 128, E), 256, 0, stream>>>(xb, w_gu, counts, offs,
                                                             bucket_tok, actR);
  routed_gemm2<<<dim3(H / 64, T / 128, E), 256, 0, stream>>>(actR, w_dn, counts, offs,
                                                             bucket_tok, bucket_w, out);
}
